// SpatialAM3D_Module_45183055954276
// MI455X (gfx1250) — matrix operand order repaired and checked by reading, not compiled, not run on hardware
//
#include <hip/hip_runtime.h>
#include <hip/hip_bf16.h>

#define B_   2
#define CIN  64
#define NSP  4096
#define CK   32
#define CV   64
#define COUT 64
#define EPSB 1e-5f
#define TK   64              // keys per attention tile
#define NIT  (NSP / TK)      // 64 key-tile iterations
// Q == K (key_query_same): pre-scale K by 32^(-1/4) so S = KK^T / sqrt(32).
#define KSCALE 0.42044820762685725f

typedef __attribute__((ext_vector_type(16))) _Float16 v16h;
typedef __attribute__((ext_vector_type(8)))  _Float16 v8h;
typedef __attribute__((ext_vector_type(8)))  float    v8f;
typedef __attribute__((__vector_size__(16))) int      v4i_;

// --- CDNA5 async global->LDS loads (ASYNCcnt), with safe fallback ----------
#if defined(__has_builtin)
#if __has_builtin(__builtin_amdgcn_global_load_async_to_lds_b128) && \
    __has_builtin(__builtin_amdgcn_s_wait_asynccnt)
#define HAVE_ASYNC_LDS 1
#endif
#endif
#ifndef HAVE_ASYNC_LDS
#define HAVE_ASYNC_LDS 0
#endif

__device__ __forceinline__ v8f wmma16x16x32(v16h a, v16h b, v8f c) {
    // (neg_a, A, neg_b, B, c_mod, C, reuse_a, reuse_b)
    return __builtin_amdgcn_wmma_f32_16x16x32_f16(false, a, false, b, (short)0, c,
                                                  false, false);
}

// ---- DPP16 butterfly reductions within each 16-lane half (no LDS, no waits)
template <int CTRL>
__device__ __forceinline__ float dpp_maxstep(float x) {
    int p = __builtin_amdgcn_update_dpp(0, __builtin_bit_cast(int, x), CTRL, 0xF,
                                        0xF, true);
    return fmaxf(x, __builtin_bit_cast(float, p));
}
template <int CTRL>
__device__ __forceinline__ float dpp_addstep(float x) {
    int p = __builtin_amdgcn_update_dpp(0, __builtin_bit_cast(int, x), CTRL, 0xF,
                                        0xF, true);
    return x + __builtin_bit_cast(float, p);
}
__device__ __forceinline__ float red_max16(float x) {
    x = dpp_maxstep<0xB1>(x);    // quad_perm(1,0,3,2)  : xor 1
    x = dpp_maxstep<0x4E>(x);    // quad_perm(2,3,0,1)  : xor 2
    x = dpp_maxstep<0x141>(x);   // row_half_mirror     : cross-quad in 8
    x = dpp_maxstep<0x140>(x);   // row_mirror          : cross-8 in 16
    return x;
}
__device__ __forceinline__ float red_add16(float x) {
    x = dpp_addstep<0xB1>(x);
    x = dpp_addstep<0x4E>(x);
    x = dpp_addstep<0x141>(x);
    x = dpp_addstep<0x140>(x);
    return x;
}

// f16 A-fragment (16x32 MxK) from a row-major buffer: lane holds row L%16,
// elements = K=[hi*16, hi*16+16) contiguous.
__device__ __forceinline__ v16h load_afrag(const _Float16* row_base, int hi) {
    v8h a0 = *(const v8h*)(row_base + hi * 16);
    v8h a1 = *(const v8h*)(row_base + hi * 16 + 8);
    v16h r;
#pragma unroll
    for (int i = 0; i < 8; ++i) { r[i] = a0[i]; r[i + 8] = a1[i]; }
    return r;
}

// ---------------------------------------------------------------------------
// Kernel 1: K projection + BatchNorm (pre-scaled by 32^-1/4)
//              -> kf16[B][N][CK] (f16, position-major)
//           V projection + bias -> vf16[B][CV][N] (f16, channel-major)
// ---------------------------------------------------------------------------
__global__ __launch_bounds__(256) void proj_kernel(
    const float* __restrict__ x, const float* __restrict__ key_w,
    const float* __restrict__ bng, const float* __restrict__ bnb,
    const float* __restrict__ bnm, const float* __restrict__ bnv,
    const float* __restrict__ value_w, const float* __restrict__ value_b,
    _Float16* __restrict__ kf16, _Float16* __restrict__ vf16) {
    __shared__ float s_kw[CK * CIN];
    __shared__ float s_vw[CV * CIN];
    __shared__ float s_scale[CK], s_shift[CK], s_vb[CV];

    const int tid = threadIdx.x;
    for (int i = tid; i < CK * CIN; i += 256) s_kw[i] = key_w[i];
    for (int i = tid; i < CV * CIN; i += 256) s_vw[i] = value_w[i];
    if (tid < CK) {
        float sc = bng[tid] * rsqrtf(bnv[tid] + EPSB);
        // fold the attention logit scale into the BN affine:
        s_scale[tid] = sc * KSCALE;
        s_shift[tid] = (bnb[tid] - bnm[tid] * sc) * KSCALE;
    }
    if (tid < CV) s_vb[tid] = value_b[tid];
    __syncthreads();

    const int g = blockIdx.x * 256 + tid;
    const int b = g >> 12;
    const int n = g & (NSP - 1);

    const float* xb = x + (size_t)b * CIN * NSP + n;
    float xr[CIN];
#pragma unroll
    for (int c = 0; c < CIN; ++c) xr[c] = xb[(size_t)c * NSP];

    {   // K with BN fused; weights read as float4 (ds_load_b128)
        float acc[CK];
#pragma unroll
        for (int o = 0; o < CK; ++o) acc[o] = 0.f;
        for (int c = 0; c < CIN; c += 4) {
#pragma unroll
            for (int o = 0; o < CK; ++o) {
                float4 w = *(const float4*)&s_kw[o * CIN + c];
                acc[o] = fmaf(w.x, xr[c + 0], acc[o]);
                acc[o] = fmaf(w.y, xr[c + 1], acc[o]);
                acc[o] = fmaf(w.z, xr[c + 2], acc[o]);
                acc[o] = fmaf(w.w, xr[c + 3], acc[o]);
            }
        }
        _Float16* kd = kf16 + (size_t)(b * NSP + n) * CK;
#pragma unroll
        for (int o = 0; o < CK; ++o)
            kd[o] = (_Float16)(acc[o] * s_scale[o] + s_shift[o]);
    }

#pragma unroll
    for (int half = 0; half < 2; ++half) {  // V in two 32-channel chunks
        float acc[32];
#pragma unroll
        for (int o = 0; o < 32; ++o) acc[o] = 0.f;
        for (int c = 0; c < CIN; c += 4) {
#pragma unroll
            for (int o = 0; o < 32; ++o) {
                float4 w = *(const float4*)&s_vw[(half * 32 + o) * CIN + c];
                acc[o] = fmaf(w.x, xr[c + 0], acc[o]);
                acc[o] = fmaf(w.y, xr[c + 1], acc[o]);
                acc[o] = fmaf(w.z, xr[c + 2], acc[o]);
                acc[o] = fmaf(w.w, xr[c + 3], acc[o]);
            }
        }
#pragma unroll
        for (int o = 0; o < 32; ++o) {
            int oo = half * 32 + o;
            vf16[(size_t)(b * CV + oo) * NSP + n] = (_Float16)(acc[o] + s_vb[oo]);
        }
    }
}

// ---------------------------------------------------------------------------
// Kernel 2: fused flash attention, 64-query block / 64-key tiles, ping-pong
// LDS with async global->LDS staging, DPP softmax reductions.  12 WMMAs/iter.
// ---------------------------------------------------------------------------
__global__ __launch_bounds__(128) void attn_kernel(
    const _Float16* __restrict__ kf16, const _Float16* __restrict__ vf16,
    float* __restrict__ ctx) {
    __shared__ __align__(32) _Float16 Kt[2][TK * CK];     // [buf][key][ch]  4KB
    __shared__ __align__(32) _Float16 Vt[2][CV * TK];     // [buf][ch][key]  8KB
    __shared__ __align__(32) _Float16 Pt[4][16 * TK];     // per-wave P      2KB

    const int tid = threadIdx.x;
    const int wv  = tid >> 5;
    const int ln  = tid & 31;
    const int r   = ln & 15;
    const int hi  = ln >> 4;
    const int b   = blockIdx.x >> 6;
    const int qb  = (blockIdx.x & 63) * 64;

    const _Float16* kglob = kf16 + (size_t)b * NSP * CK;
    const _Float16* vglob = vf16 + (size_t)b * CV * NSP;
    const int ch_v = tid >> 1, seg_v = tid & 1;

    // Q fragment: resident in registers the whole kernel (already pre-scaled).
    const int qrow = qb + wv * 16 + r;
    v16h aq = load_afrag(kglob + (size_t)qrow * CK, hi);

    v8f o0 = {}, o1 = {}, o2 = {}, o3 = {};
    float m[8], l[8];
#pragma unroll
    for (int i = 0; i < 8; ++i) { m[i] = -1e30f; l[i] = 0.f; }

#if HAVE_ASYNC_LDS
    // ---- fire-and-forget async tile staging (tracked by ASYNCcnt) ---------
#define STAGE_TILE(kt0, buf)                                                   \
    do {                                                                       \
        const uint4* ks = (const uint4*)(kglob + (size_t)(kt0)*CK) + tid;      \
        uint4* kd = (uint4*)Kt[buf] + tid;                                     \
        __builtin_amdgcn_global_load_async_to_lds_b128((v4i_*)ks, (v4i_*)kd,   \
                                                       0, 0);                  \
        __builtin_amdgcn_global_load_async_to_lds_b128((v4i_*)ks, (v4i_*)kd,   \
                                                       2048, 0);               \
        const uint4* vs =                                                      \
            (const uint4*)(vglob + (size_t)ch_v * NSP + (kt0) + seg_v * 32);   \
        uint4* vd = (uint4*)(Vt[buf] + ch_v * TK + seg_v * 32);                \
        __builtin_amdgcn_global_load_async_to_lds_b128((v4i_*)vs, (v4i_*)vd,   \
                                                       0, 0);                  \
        __builtin_amdgcn_global_load_async_to_lds_b128((v4i_*)vs, (v4i_*)vd,   \
                                                       16, 0);                 \
        __builtin_amdgcn_global_load_async_to_lds_b128((v4i_*)vs, (v4i_*)vd,   \
                                                       32, 0);                 \
        __builtin_amdgcn_global_load_async_to_lds_b128((v4i_*)vs, (v4i_*)vd,   \
                                                       48, 0);                 \
    } while (0)
#define STAGE_WAIT() __builtin_amdgcn_s_wait_asynccnt(0)
#else
    // ---- fallback: register-staged copies ---------------------------------
    uint4 rk0, rk1, rv0, rv1, rv2, rv3;
#define STAGE_TILE(kt0, buf)                                                   \
    do {                                                                       \
        const uint4* ks = (const uint4*)(kglob + (size_t)(kt0)*CK);            \
        rk0 = ks[tid];                                                         \
        rk1 = ks[tid + 128];                                                   \
        const uint4* vs =                                                      \
            (const uint4*)(vglob + (size_t)ch_v * NSP + (kt0) + seg_v * 32);   \
        rv0 = vs[0]; rv1 = vs[1]; rv2 = vs[2]; rv3 = vs[3];                    \
        uint4* kd = (uint4*)Kt[buf];                                           \
        kd[tid] = rk0;                                                         \
        kd[tid + 128] = rk1;                                                   \
        uint4* vd = (uint4*)(Vt[buf] + ch_v * TK + seg_v * 32);                \
        vd[0] = rv0; vd[1] = rv1; vd[2] = rv2; vd[3] = rv3;                    \
    } while (0)
#define STAGE_WAIT() ((void)0)
#endif

    STAGE_TILE(0, 0);
    STAGE_WAIT();
    __syncthreads();

    for (int kt = 0; kt < NIT; ++kt) {
        const int cur = kt & 1;
        if (kt + 1 < NIT) STAGE_TILE((kt + 1) * TK, cur ^ 1);
        if (kt + 2 < NIT)  // prefetch 2 tiles ahead into L2 (global_prefetch_b8)
            __builtin_prefetch(kglob + (size_t)(kt + 2) * TK * CK + tid * 32, 0, 1);

        // --- S = Q K^T (pre-scaled) : 4 column tiles ---
        v8f s0 = {}, s1 = {}, s2 = {}, s3 = {};
        {
            const _Float16* kb = Kt[cur];
            v16h bk0 = *(const v16h*)(kb + (0 * 16 + r) * CK + hi * 16);
            v16h bk1 = *(const v16h*)(kb + (1 * 16 + r) * CK + hi * 16);
            v16h bk2 = *(const v16h*)(kb + (2 * 16 + r) * CK + hi * 16);
            v16h bk3 = *(const v16h*)(kb + (3 * 16 + r) * CK + hi * 16);
            s0 = wmma16x16x32(aq, bk0, s0);
            s1 = wmma16x16x32(aq, bk1, s1);
            s2 = wmma16x16x32(aq, bk2, s2);
            s3 = wmma16x16x32(aq, bk3, s3);
        }

        // --- online softmax (DPP reductions, pure VALU) ---
#pragma unroll
        for (int gI = 0; gI < 8; ++gI) {
            float a0 = s0[gI], a1 = s1[gI], a2 = s2[gI], a3 = s3[gI];
            float mx = red_max16(fmaxf(fmaxf(a0, a1), fmaxf(a2, a3)));
            float mn = fmaxf(m[gI], mx);
            float p0 = __expf(a0 - mn), p1 = __expf(a1 - mn);
            float p2 = __expf(a2 - mn), p3 = __expf(a3 - mn);
            float corr = __expf(m[gI] - mn);
            float rs = red_add16((p0 + p1) + (p2 + p3));
            l[gI] = l[gI] * corr + rs;
            m[gI] = mn;
            o0[gI] *= corr; o1[gI] *= corr; o2[gI] *= corr; o3[gI] *= corr;
            int row = gI + 8 * hi;
            _Float16* pr = &Pt[wv][row * TK];
            pr[0 * 16 + r] = (_Float16)p0;
            pr[1 * 16 + r] = (_Float16)p1;
            pr[2 * 16 + r] = (_Float16)p2;
            pr[3 * 16 + r] = (_Float16)p3;
        }

        // --- O += P V : 4 channel tiles x 2 key chunks ---
        {
            v16h ap0 = load_afrag(&Pt[wv][r * TK], hi);        // keys 0..31
            v16h ap1 = load_afrag(&Pt[wv][r * TK + 32], hi);   // keys 32..63
            const _Float16* vb = Vt[cur];
            v16h b0l = *(const v16h*)(vb + (0 * 16 + r) * TK + hi * 16);
            v16h b1l = *(const v16h*)(vb + (1 * 16 + r) * TK + hi * 16);
            v16h b2l = *(const v16h*)(vb + (2 * 16 + r) * TK + hi * 16);
            v16h b3l = *(const v16h*)(vb + (3 * 16 + r) * TK + hi * 16);
            o0 = wmma16x16x32(ap0, b0l, o0);
            o1 = wmma16x16x32(ap0, b1l, o1);
            o2 = wmma16x16x32(ap0, b2l, o2);
            o3 = wmma16x16x32(ap0, b3l, o3);
            v16h b0h = *(const v16h*)(vb + (0 * 16 + r) * TK + 32 + hi * 16);
            v16h b1h = *(const v16h*)(vb + (1 * 16 + r) * TK + 32 + hi * 16);
            v16h b2h = *(const v16h*)(vb + (2 * 16 + r) * TK + 32 + hi * 16);
            v16h b3h = *(const v16h*)(vb + (3 * 16 + r) * TK + 32 + hi * 16);
            o0 = wmma16x16x32(ap1, b0h, o0);
            o1 = wmma16x16x32(ap1, b1h, o1);
            o2 = wmma16x16x32(ap1, b2h, o2);
            o3 = wmma16x16x32(ap1, b3h, o3);
        }

        STAGE_WAIT();       // next tile fully in LDS (ASYNCcnt==0)
        __syncthreads();
    }

    // --- normalize + write ctx [B][N][CV] fp32 ---
#pragma unroll
    for (int gI = 0; gI < 8; ++gI) {
        float inv = 1.0f / l[gI];
        int q = qb + wv * 16 + gI + 8 * hi;
        float* cd = ctx + (size_t)(b * NSP + q) * CV;
        cd[0 * 16 + r] = o0[gI] * inv;
        cd[1 * 16 + r] = o1[gI] * inv;
        cd[2 * 16 + r] = o2[gI] * inv;
        cd[3 * 16 + r] = o3[gI] * inv;
    }
#undef STAGE_TILE
#undef STAGE_WAIT
}

// ---------------------------------------------------------------------------
// Kernel 3: out = gamma * (W ctx + Wb) + x
// ---------------------------------------------------------------------------
__global__ __launch_bounds__(256) void out_kernel(
    const float* __restrict__ ctx, const float* __restrict__ Ww,
    const float* __restrict__ Wb, const float* __restrict__ gamma,
    const float* __restrict__ x, float* __restrict__ out) {
    __shared__ float s_w[COUT * CV];
    __shared__ float s_b[COUT];
    const int tid = threadIdx.x;
    for (int i = tid; i < COUT * CV; i += 256) s_w[i] = Ww[i];
    if (tid < COUT) s_b[tid] = Wb[tid];
    __syncthreads();

    const int g = blockIdx.x * 256 + tid;
    const int b = g >> 12;
    const int n = g & (NSP - 1);

    float cr[CV];
    const float4* cp = (const float4*)(ctx + (size_t)(b * NSP + n) * CV);
#pragma unroll
    for (int v = 0; v < CV / 4; ++v) {
        float4 c4 = cp[v];
        cr[4 * v + 0] = c4.x; cr[4 * v + 1] = c4.y;
        cr[4 * v + 2] = c4.z; cr[4 * v + 3] = c4.w;
    }
    const float gm = gamma[0];

#pragma unroll
    for (int oc4 = 0; oc4 < 4; ++oc4) {
        float acc[16];
#pragma unroll
        for (int o = 0; o < 16; ++o) acc[o] = 0.f;
        for (int v = 0; v < CV; v += 4) {
#pragma unroll
            for (int o = 0; o < 16; ++o) {
                float4 w = *(const float4*)&s_w[(oc4 * 16 + o) * CV + v];
                acc[o] = fmaf(w.x, cr[v + 0], acc[o]);
                acc[o] = fmaf(w.y, cr[v + 1], acc[o]);
                acc[o] = fmaf(w.z, cr[v + 2], acc[o]);
                acc[o] = fmaf(w.w, cr[v + 3], acc[o]);
            }
        }
#pragma unroll
        for (int o = 0; o < 16; ++o) {
            int oo = oc4 * 16 + o;
            size_t idx = (size_t)(b * COUT + oo) * NSP + n;
            out[idx] = gm * (acc[o] + s_b[oo]) + x[idx];
        }
    }
}

extern "C" void kernel_launch(void* const* d_in, const int* in_sizes, int n_in,
                              void* d_out, int out_size, void* d_ws, size_t ws_size,
                              hipStream_t stream) {
    (void)in_sizes; (void)n_in; (void)out_size; (void)ws_size;
    const float* x       = (const float*)d_in[0];
    const float* key_w   = (const float*)d_in[1];
    const float* bng     = (const float*)d_in[2];
    const float* bnb     = (const float*)d_in[3];
    const float* bnm     = (const float*)d_in[4];
    const float* bnv     = (const float*)d_in[5];
    const float* value_w = (const float*)d_in[6];
    const float* value_b = (const float*)d_in[7];
    const float* W_w     = (const float*)d_in[8];
    const float* W_b     = (const float*)d_in[9];
    const float* gamma   = (const float*)d_in[10];

    // Workspace layout: kf16 (512KB) | vf16 (1MB) | ctx (2MB)
    _Float16* kf16 = (_Float16*)d_ws;
    _Float16* vf16 = kf16 + (size_t)B_ * NSP * CK;
    float*    ctx  = (float*)(vf16 + (size_t)B_ * CV * NSP);

    proj_kernel<<<(B_ * NSP) / 256, 256, 0, stream>>>(
        x, key_w, bng, bnb, bnm, bnv, value_w, value_b, kf16, vf16);
    attn_kernel<<<B_ * (NSP / 64), 128, 0, stream>>>(kf16, vf16, ctx);
    out_kernel<<<(B_ * NSP) / 256, 256, 0, stream>>>(
        ctx, W_w, W_b, gamma, x, (float*)d_out);
}
